// VQVAE_Simp_9242769622323
// MI455X (gfx1250) — matrix operand order repaired and checked by reading, not compiled, not run on hardware
//
#include <hip/hip_runtime.h>

// ---------------------------------------------------------------------------
// VQ-VAE forward for MI455X (gfx1250): all convs / codebook distances as
// implicit GEMMs on v_wmma_f32_16x16x32_bf16 (bf16 in, f32 accumulate).
// Weights/codebook staged to LDS via Tensor Data Mover (TDM) when available.
// ---------------------------------------------------------------------------

typedef __attribute__((ext_vector_type(16))) __bf16 bf16x16;
typedef __attribute__((ext_vector_type(8)))  float  fx8;

union ABfrag { bf16x16 v; uint4 q[2]; };

#if __has_builtin(__builtin_amdgcn_tensor_load_to_lds) && \
    __has_builtin(__builtin_amdgcn_s_wait_tensorcnt)
#define HAVE_TDM 1
#else
#define HAVE_TDM 0
#endif

#if HAVE_TDM
typedef __attribute__((ext_vector_type(4))) unsigned int u32x4;
typedef __attribute__((ext_vector_type(8))) int          i32x8;
typedef __attribute__((ext_vector_type(4))) int          i32x4;

// 2D tile load: rows x rowBytes from global (row stride strideB bytes) -> LDS.
// D# built per CDNA5 ISA 8.3/8.4 (data_size=1B, type=2, no padding/iterate).
// This toolchain declares the 6-arg builtin (4 SGPR groups + spare + cpol).
__device__ __forceinline__ void tdm_load_2d(const void* g, void* l,
                                            unsigned rowBytes, unsigned rows,
                                            unsigned long long strideB) {
  unsigned long long ga = (unsigned long long)g;
  unsigned ldsOff = (unsigned)(unsigned long long)l;
  u32x4 g0 = {0u, 0u, 0u, 0u};
  g0[0] = 1u;                                   // count=1 (valid user D#)
  g0[1] = ldsOff;                               // lds_addr
  g0[2] = (unsigned)ga;                         // global_addr[31:0]
  g0[3] = (unsigned)((ga >> 32) & 0x01FFFFFFull) | (2u << 30);  // [56:32]|type=2
  unsigned long long td0 = strideB;             // tensor_dim0 (1B units)
  unsigned td1 = rows;                          // tensor_dim1
  i32x8 g1 = {0, 0, 0, 0, 0, 0, 0, 0};
  g1[1] = (int)((td0 & 0xFFFFull) << 16);                       // dim0[15:0]
  g1[2] = (int)(((td0 >> 16) & 0xFFFFull) | ((unsigned long long)(td1 & 0xFFFFu) << 16));
  g1[3] = (int)(((td1 >> 16) & 0xFFFFu) | (rowBytes << 16));    // tile_dim0
  g1[4] = (int)rows;                                            // tile_dim1
  g1[5] = (int)(strideB & 0xFFFFFFFFull);                       // dim0_stride lo
  g1[6] = (int)((strideB >> 32) & 0xFFFFull);                   // dim0_stride hi
  i32x4 z4 = {0, 0, 0, 0};
  i32x8 z8 = {0, 0, 0, 0, 0, 0, 0, 0};
  __builtin_amdgcn_tensor_load_to_lds(g0, g1, z4, z4, z8, 0);
}
#endif

__device__ __forceinline__ fx8 wmma_bf16(bf16x16 a, bf16x16 b, fx8 c) {
  return __builtin_amdgcn_wmma_f32_16x16x32_bf16(false, a, false, b,
                                                 (short)0, c, false, false);
}

__device__ __forceinline__ unsigned relu_pk(unsigned v) {
  unsigned s = (v >> 15) & 0x00010001u;   // sign bit per bf16 half
  return v & ~(s * 0xFFFFu);              // zero negative halves
}
__device__ __forceinline__ uint4 relu_pk4(uint4 v) {
  v.x = relu_pk(v.x); v.y = relu_pk(v.y);
  v.z = relu_pk(v.z); v.w = relu_pk(v.w);
  return v;
}
__device__ __forceinline__ unsigned short f2bf(float f) {
  unsigned u = __float_as_uint(f);
  u = u + 0x7FFFu + ((u >> 16) & 1u);     // round-to-nearest-even
  return (unsigned short)(u >> 16);
}
__device__ __forceinline__ float bf2f(unsigned short h) {
  return __uint_as_float(((unsigned)h) << 16);
}

enum {
  F_RELU_IN    = 1,
  F_RELU_OUT   = 2,
  F_RESID      = 4,
  F_STORE_BF16 = 8,
  F_STORE_ZE   = 16,
  F_FINAL      = 32
};

// ---------------------------------------------------------------------------
// Implicit-GEMM conv. One wave: 16 pixels x Cout. Weight K-strips (Cout x 32
// bf16) are double-buffered in LDS (TDM prefetch overlapped with compute).
// All div/mod removed: W/16, H are powers of two; tap/KW via magic multiply.
// ---------------------------------------------------------------------------
__global__ __launch_bounds__(256) void conv_wmma_kernel(
    const unsigned short* __restrict__ in, int inHs, int inCp, int cShift,
    const unsigned short* __restrict__ Wp,
    const float* __restrict__ scale, const float* __restrict__ shift,
    unsigned short* __restrict__ outb, int outHs, int outHalo, int outCp,
    const unsigned short* __restrict__ resid,
    float* __restrict__ zef, unsigned short* __restrict__ zeb,
    const float* __restrict__ xin, float* __restrict__ xrec,
    float* __restrict__ rsum,
    int H, int W, int Cout, int K, int KW, int kwMagic, int stride, int inOff,
    int twShift, int hShift, int flags)
{
  __shared__ unsigned short sW[2][2048];      // 2 x (Cout<=64 x 32 bf16)
  const int tid  = threadIdx.x;
  const int lane = tid & 31;
  long wid = (long)blockIdx.x * 8 + (tid >> 5);
  int  tx  = (int)(wid & (long)((1 << twShift) - 1));
  long t   = wid >> twShift;
  int  oy  = (int)(t & (long)((1 << hShift) - 1));
  long bb  = t >> hShift;
  const bool active = (bb < 32);
  int b = active ? (int)bb : 0;
  if (!active) { oy = 0; tx = 0; }
  const int ox0 = tx << 4;
  const int m   = lane & 15;
  const int hi  = lane >> 4;
  const int nTiles = Cout >> 4;
  const int cMask  = inCp - 1;
  const int elems  = Cout << 5;              // strip elements

  // --- strip staging (TDM or cooperative load+ds_store fallback) ---
  auto stage_issue = [&](int buf, int k0) {
#if HAVE_TDM
    if (tid < 32)
      tdm_load_2d(Wp + k0, &sW[buf][0], 64u, (unsigned)Cout,
                  (unsigned long long)K * 2ull);
#else
    int e = tid << 3;
    if (e < elems) {
      int n = e >> 5, kk = e & 31;
      *(uint4*)&sW[buf][e] = *(const uint4*)(Wp + (long)n * K + k0 + kk);
    }
#endif
  };
  auto stage_wait = [&]() {
#if HAVE_TDM
    if (tid < 32) __builtin_amdgcn_s_wait_tensorcnt(0);
#endif
  };

  const fx8 z8f = {0.f,0.f,0.f,0.f,0.f,0.f,0.f,0.f};
  fx8 acc[4];
#pragma unroll
  for (int i = 0; i < 4; ++i) acc[i] = z8f;

  stage_issue(0, 0);
  stage_wait();
  __syncthreads();

  int cb = 0;
  for (int k0 = 0; k0 < K; k0 += 32) {
    if (k0 + 32 < K) stage_issue(cb ^ 1, k0 + 32);   // overlap next strip

    // ---- A fragment: 16x32 bf16, ISA lane layout ----
    ABfrag af;
#pragma unroll
    for (int g = 0; g < 2; ++g) {
      int kb  = k0 + g * 16 + hi * 8;
      int tap = kb >> cShift;
      int c   = kb & cMask;
      int ty  = (tap * kwMagic) >> 8;       // tap / KW
      int txp = tap - ty * KW;
      int py  = oy * stride + ty + inOff;
      int px  = (ox0 + m) * stride + txp + inOff;
      const unsigned short* p =
          in + (((long)b * inHs + py) * inHs + px) * inCp + c;
      __builtin_prefetch(p + inCp, 0, 1);   // global_prefetch_b8
      uint4 v = *(const uint4*)p;
      if (flags & F_RELU_IN) v = relu_pk4(v);
      af.q[g] = v;
    }
    // ---- B fragments from LDS strip + WMMA per 16-wide N tile ----
#pragma unroll
    for (int nt = 0; nt < 4; ++nt) {
      if (nt < nTiles) {
        const unsigned short* ws = &sW[cb][((nt << 4) + m) * 32 + hi * 8];
        ABfrag bfr;
        bfr.q[0] = *(const uint4*)ws;
        bfr.q[1] = *(const uint4*)(ws + 16);
        acc[nt] = wmma_bf16(af.v, bfr.v, acc[nt]);
      }
    }
    stage_wait();
    __syncthreads();
    cb ^= 1;
  }

  // ---- epilogue: scale/shift (BN+bias), relu, residual, stores ----
  float lsum = 0.f;
  long  orow = ((long)b * outHs + oy + outHalo) * outHs;
#pragma unroll
  for (int nt = 0; nt < 4; ++nt) {
    if (nt < nTiles) {
      int   ch = (nt << 4) + m;
      float sc = scale[ch], sh = shift[ch];
#pragma unroll
      for (int r = 0; r < 8; ++r) {
        int   px = ox0 + (hi ? 8 + r : r);
        float v  = acc[nt][r] * sc + sh;
        if (flags & F_RESID)
          v += bf2f(resid[(orow + px + outHalo) * outCp + ch]);
        if (flags & F_RELU_OUT) v = fmaxf(v, 0.f);
        if (active && (flags & F_STORE_BF16))
          outb[(orow + px + outHalo) * outCp + ch] = f2bf(v);
        if (active && (flags & F_STORE_ZE)) {
          long row = ((long)b * H + oy) * W + px;
          zef[row * 64 + ch] = v;
          zeb[row * 64 + ch] = f2bf(v);
        }
        if (flags & F_FINAL) {
          float yv = tanhf(v);
          if (active && ch < 15) {
            long o = (((long)b * 15 + ch) * H + oy) * W + px;
            xrec[o] = yv;
            float d = yv - xin[o];
            lsum += d * d;
          }
        }
      }
    }
  }
  if (active && (flags & F_FINAL)) atomicAdd(rsum, lsum);
}

// ---------------------------------------------------------------------------
// VQ argmin: dist = ||c||^2 - 2*ze.c (row term constant). The full 256 KB
// codebook is staged into WGP LDS (320 KB) once, reused by all 8 waves.
// ---------------------------------------------------------------------------
__global__ __launch_bounds__(256) void vq_argmin_kernel(
    const unsigned short* __restrict__ zeb,
    const unsigned short* __restrict__ cbb,
    const float* __restrict__ cnorm, int* __restrict__ idxbuf)
{
  extern __shared__ unsigned short sCB[];   // 2048 x 64 bf16 = 256 KB
  const int tid  = threadIdx.x;
  const int lane = tid & 31;
#if HAVE_TDM
  if (tid < 32) {
    tdm_load_2d(cbb, sCB, 128u, 2048u, 128ull);
    __builtin_amdgcn_s_wait_tensorcnt(0);
  }
#else
  for (int e = tid << 3; e < 2048 * 64; e += 256 * 8)
    *(uint4*)&sCB[e] = *(const uint4*)(cbb + e);
#endif
  __syncthreads();

  long wid = (long)blockIdx.x * 8 + (tid >> 5);
  long m0  = wid << 4;                      // 16 rows (pixels) per wave
  int  m   = lane & 15;
  int  hi  = lane >> 4;

  const unsigned short* zp = zeb + (m0 + m) * 64;
  ABfrag a0, a1;
  a0.q[0] = *(const uint4*)(zp + hi * 8);
  a0.q[1] = *(const uint4*)(zp + 16 + hi * 8);
  a1.q[0] = *(const uint4*)(zp + 32 + hi * 8);
  a1.q[1] = *(const uint4*)(zp + 48 + hi * 8);

  float best[8]; int bidx[8];
#pragma unroll
  for (int r = 0; r < 8; ++r) { best[r] = 3.4e38f; bidx[r] = 0; }
  const fx8 z8f = {0.f,0.f,0.f,0.f,0.f,0.f,0.f,0.f};

  for (int nt = 0; nt < 128; ++nt) {
    int n = (nt << 4) + m;
    const unsigned short* cp = sCB + n * 64 + hi * 8;
    ABfrag b0, b1;
    b0.q[0] = *(const uint4*)cp;        b0.q[1] = *(const uint4*)(cp + 16);
    b1.q[0] = *(const uint4*)(cp + 32); b1.q[1] = *(const uint4*)(cp + 48);
    fx8 acc = z8f;
    acc = wmma_bf16(a0.v, b0.v, acc);
    acc = wmma_bf16(a1.v, b1.v, acc);
    float cn = cnorm[n];
#pragma unroll
    for (int r = 0; r < 8; ++r) {
      float d = cn - 2.f * acc[r];
      if (d < best[r]) { best[r] = d; bidx[r] = n; }
    }
  }
#pragma unroll
  for (int r = 0; r < 8; ++r) {
    float bv = best[r]; int bi = bidx[r];
#pragma unroll
    for (int off = 8; off > 0; off >>= 1) {
      float ov = __shfl_xor(bv, off, 16);
      int   oi = __shfl_xor(bi, off, 16);
      if (ov < bv || (ov == bv && oi < bi)) { bv = ov; bi = oi; }
    }
    if (m == 0) idxbuf[m0 + hi * 8 + r] = bi;
  }
}

// quantized gather: write q (bf16, halo-1 NHWC), counts, sum((q-ze)^2)
__global__ __launch_bounds__(256) void vq_gather_kernel(
    const int* __restrict__ idxbuf, const float* __restrict__ cb,
    const float* __restrict__ zef, unsigned short* __restrict__ qbuf,
    float* __restrict__ counts, float* __restrict__ vq_sum)
{
  __shared__ float sm[256];
  int t = threadIdx.x;
  int c = t & 63, rl = t >> 6;
  long row = (long)blockIdx.x * 4 + rl;
  int  id  = idxbuf[row];
  float qv = cb[(long)id * 64 + c];
  float zv = zef[row * 64 + c];
  float d  = qv - zv;
  int b = (int)(row >> 10), y = (int)((row >> 5) & 31), x = (int)(row & 31);
  qbuf[(((long)b * 34 + y + 1) * 34 + x + 1) * 64 + c] = f2bf(qv);
  if (c == 0) atomicAdd(&counts[id], 1.f);
  sm[t] = d * d;
  __syncthreads();
  for (int s = 128; s > 0; s >>= 1) {
    if (t < s) sm[t] += sm[t + s];
    __syncthreads();
  }
  if (t == 0) atomicAdd(vq_sum, sm[0]);
}

// ---------------------------------------------------------------------------
// packing / utility kernels
// ---------------------------------------------------------------------------
__global__ void zero_kernel(unsigned int* __restrict__ p, long n) {
  long i = (long)blockIdx.x * blockDim.x + threadIdx.x;
  long s = (long)gridDim.x * blockDim.x;
  for (; i < n; i += s) p[i] = 0u;
}

__global__ void pack_input_kernel(const float* __restrict__ x,
                                  unsigned short* __restrict__ out) {
  long i = (long)blockIdx.x * blockDim.x + threadIdx.x;
  if (i >= (long)32 * 256 * 256 * 16) return;
  int c = (int)(i & 15); long p = i >> 4;
  int xx = (int)(p & 255); p >>= 8;
  int yy = (int)(p & 255); p >>= 8;
  int b  = (int)p;
  float v = 0.f;
  if (c < 15) v = x[(((long)b * 15 + c) * 256 + yy) * 256 + xx];
  out[(((long)b * 258 + yy + 1) * 258 + xx + 1) * 16 + c] = f2bf(v);
}

// OIHW (or IOHW flipped, for conv_t) f32 -> [Npad][K] bf16, K = KH*KW*Cp
__global__ void pack_w_kernel(const float* __restrict__ w,
                              unsigned short* __restrict__ out,
                              int O, int I, int KH, int KW, int Cp, int Npad,
                              int tflip) {
  long K = (long)KH * KW * Cp;
  long total = (long)Npad * K;
  long i = (long)blockIdx.x * blockDim.x + threadIdx.x;
  if (i >= total) return;
  int n = (int)(i / K); long k = i % K;
  int tap = (int)(k / Cp), c = (int)(k % Cp);
  int ky = tap / KW, kx = tap % KW;
  float v = 0.f;
  if (n < O && c < I) {
    if (tflip) v = w[(((long)c * O + n) * KH + (KH - 1 - ky)) * KW + (KW - 1 - kx)];
    else       v = w[(((long)n * I + c) * KH + ky) * KW + kx];
  }
  out[i] = f2bf(v);
}

// fold BN+bias -> per-channel scale/shift (scale=0 for padded channels)
__global__ void pack_ss_kernel(const float* __restrict__ bias,
                               const float* __restrict__ bn,
                               float* __restrict__ scale,
                               float* __restrict__ shift, int C, int Cp) {
  int c = blockIdx.x * blockDim.x + threadIdx.x;
  if (c >= Cp) return;
  if (c >= C) { scale[c] = 0.f; shift[c] = 0.f; return; }
  float b = bias ? bias[c] : 0.f;
  if (bn) {
    float g = bn[0 * C + c], be = bn[1 * C + c];
    float mn = bn[2 * C + c], vr = bn[3 * C + c];
    float s = g * rsqrtf(vr + 1e-5f);
    scale[c] = s; shift[c] = (b - mn) * s + be;
  } else {
    scale[c] = 1.f; shift[c] = b;
  }
}

__global__ void pack_cb_kernel(const float* __restrict__ cb,
                               unsigned short* __restrict__ cbb,
                               float* __restrict__ cnorm) {
  int n = blockIdx.x * blockDim.x + threadIdx.x;
  if (n >= 2048) return;
  float s = 0.f;
  for (int c = 0; c < 64; ++c) {
    float v = cb[(long)n * 64 + c];
    s += v * v;
    cbb[(long)n * 64 + c] = f2bf(v);
  }
  cnorm[n] = s;
}

// zero-dilate x2 into pre-zeroed halo-2 buffer (conv_t lowering)
__global__ void upsample_kernel(const unsigned short* __restrict__ in,
                                unsigned short* __restrict__ out,
                                long total, int hShift, int inHs, int outHs,
                                int relu) {
  long i = (long)blockIdx.x * blockDim.x + threadIdx.x;
  if (i >= total) return;
  int mask = (1 << hShift) - 1;
  int cb = (int)(i & 7); long p = i >> 3;
  int x = (int)(p & mask); p >>= hShift;
  int y = (int)(p & mask);
  int b = (int)(p >> hShift);
  uint4 v = *(const uint4*)(in + (((long)b * inHs + y + 1) * inHs + x + 1) * 64 + cb * 8);
  if (relu) v = relu_pk4(v);
  *(uint4*)(out + (((long)b * outHs + 2 * y + 2) * outHs + 2 * x + 2) * 64 + cb * 8) = v;
}

__global__ void finalize_kernel(const float* __restrict__ counts,
                                const float* __restrict__ vq_sum,
                                const float* __restrict__ rsum,
                                float* __restrict__ dout, long numel) {
  __shared__ float sm[256];
  int t = threadIdx.x;
  float acc = 0.f;
  for (int i = t; i < 2048; i += 256) {
    float p = counts[i] * (1.f / 32768.f);
    acc += p * logf(p + 1e-10f);
  }
  sm[t] = acc;
  __syncthreads();
  for (int s = 128; s > 0; s >>= 1) {
    if (t < s) sm[t] += sm[t + s];
    __syncthreads();
  }
  if (t == 0) {
    float perp    = expf(-sm[0]);
    float vq_loss = 1.25f * vq_sum[0] / (32768.f * 64.f);  // e_loss==q_loss fwd
    float mse     = rsum[0] / (float)numel;
    dout[numel]     = mse + vq_loss;
    dout[numel + 1] = perp;
  }
}

// ---------------------------------------------------------------------------
// host: orchestration
// ---------------------------------------------------------------------------
extern "C" void kernel_launch(void* const* d_in, const int* in_sizes, int n_in,
                              void* d_out, int out_size, void* d_ws,
                              size_t ws_size, hipStream_t stream) {
  (void)in_sizes; (void)n_in; (void)out_size;

  const float* x        = (const float*)d_in[0];
  const float* ew1      = (const float*)d_in[1];
  const float* eb1      = (const float*)d_in[2];
  const float* ew2      = (const float*)d_in[3];
  const float* eb2      = (const float*)d_in[4];
  const float* ew3      = (const float*)d_in[5];
  const float* eb3      = (const float*)d_in[6];
  const float* enc_bn   = (const float*)d_in[7];
  const float* erw3     = (const float*)d_in[8];
  const float* erb3     = (const float*)d_in[9];
  const float* erw1     = (const float*)d_in[10];
  const float* erb1     = (const float*)d_in[11];
  const float* erbn     = (const float*)d_in[12];
  const float* eow      = (const float*)d_in[13];
  const float* eob      = (const float*)d_in[14];
  const float* codebook = (const float*)d_in[15];
  const float* diw      = (const float*)d_in[16];
  const float* dib      = (const float*)d_in[17];
  const float* drw3     = (const float*)d_in[18];
  const float* drb3     = (const float*)d_in[19];
  const float* drw1     = (const float*)d_in[20];
  const float* drb1     = (const float*)d_in[21];
  const float* drbn     = (const float*)d_in[22];
  const float* dtw1     = (const float*)d_in[23];
  const float* dtb1     = (const float*)d_in[24];
  const float* dtw2     = (const float*)d_in[25];
  const float* dtb2     = (const float*)d_in[26];
  const float* dtw3     = (const float*)d_in[27];
  const float* dtb3     = (const float*)d_in[28];
  float* dout = (float*)d_out;

  char*  base = (char*)d_ws;
  size_t cur  = 0;
  auto alloc = [&](size_t bytes) -> char* {
    char* p = base + cur;
    cur = (cur + bytes + 255) & ~(size_t)255;
    return p;
  };
  const size_t US = sizeof(unsigned short);

  // activations (bf16 NHWC, square, with halo)
  unsigned short* A0 = (unsigned short*)alloc((size_t)32*258*258*16*US);
  unsigned short* Z1 = (unsigned short*)alloc((size_t)32*130*130*64*US);
  unsigned short* Z2 = (unsigned short*)alloc((size_t)32*66*66*64*US);
  unsigned short* RA = (unsigned short*)alloc((size_t)32*34*34*64*US);
  unsigned short* RT = (unsigned short*)alloc((size_t)32*34*34*64*US);
  unsigned short* RB = (unsigned short*)alloc((size_t)32*34*34*64*US);
  unsigned short* Qb = (unsigned short*)alloc((size_t)32*34*34*64*US);
  float*          ZEF = (float*)alloc((size_t)32768*64*4);
  unsigned short* ZEB = (unsigned short*)alloc((size_t)32768*64*US);
  int*            IDX = (int*)alloc((size_t)32768*4);
  float*          CNT = (float*)alloc((size_t)2048*4);
  float*          VQS = (float*)alloc(4);
  float*          RSM = (float*)alloc(4);
  unsigned short* U1 = (unsigned short*)alloc((size_t)32*67*67*64*US);
  unsigned short* V1 = (unsigned short*)alloc((size_t)32*66*66*64*US);
  unsigned short* U2 = (unsigned short*)alloc((size_t)32*131*131*64*US);
  unsigned short* V2 = (unsigned short*)alloc((size_t)32*130*130*64*US);
  unsigned short* U3 = (unsigned short*)alloc((size_t)32*259*259*64*US);
  // weights
  unsigned short* Wc1 = (unsigned short*)alloc((size_t)64*256*US);
  unsigned short* Wc2 = (unsigned short*)alloc((size_t)64*1024*US);
  unsigned short* Wc3 = (unsigned short*)alloc((size_t)64*1024*US);
  unsigned short* WeR3[4], *WeR1[4], *WdR3[4], *WdR1[4];
  for (int i = 0; i < 4; ++i) WeR3[i] = (unsigned short*)alloc((size_t)64*576*US);
  for (int i = 0; i < 4; ++i) WeR1[i] = (unsigned short*)alloc((size_t)64*64*US);
  unsigned short* Weo = (unsigned short*)alloc((size_t)64*64*US);
  unsigned short* Wdi = (unsigned short*)alloc((size_t)64*576*US);
  for (int i = 0; i < 4; ++i) WdR3[i] = (unsigned short*)alloc((size_t)64*576*US);
  for (int i = 0; i < 4; ++i) WdR1[i] = (unsigned short*)alloc((size_t)64*64*US);
  unsigned short* Wt1 = (unsigned short*)alloc((size_t)64*1024*US);
  unsigned short* Wt2 = (unsigned short*)alloc((size_t)64*1024*US);
  unsigned short* Wt3 = (unsigned short*)alloc((size_t)16*1024*US);
  unsigned short* CBB = (unsigned short*)alloc((size_t)2048*64*US);
  float* CNORM = (float*)alloc((size_t)2048*4);
  float* SS = (float*)alloc((size_t)24*128*4);

  if (cur > ws_size) return;  // workspace too small; nothing safe to do

  auto cdiv = [](long a, long b) { return (int)((a + b - 1) / b); };

  // zero whole workspace span (halos, dilation zeros, accumulators)
  zero_kernel<<<8192, 256, 0, stream>>>((unsigned int*)base, (long)(cur / 4));

  // pack input & weights
  pack_input_kernel<<<cdiv((long)32*256*256*16, 256), 256, 0, stream>>>(x, A0);
  auto packw = [&](const float* w, unsigned short* o, int O, int I, int KH,
                   int KW, int Cp, int Npad, int tflip) {
    long tot = (long)Npad * KH * KW * Cp;
    pack_w_kernel<<<cdiv(tot, 256), 256, 0, stream>>>(w, o, O, I, KH, KW, Cp,
                                                      Npad, tflip);
  };
  packw(ew1, Wc1, 64, 15, 4, 4, 16, 64, 0);
  packw(ew2, Wc2, 64, 64, 4, 4, 64, 64, 0);
  packw(ew3, Wc3, 64, 64, 4, 4, 64, 64, 0);
  for (int i = 0; i < 4; ++i) {
    packw(erw3 + (long)i*64*64*9, WeR3[i], 64, 64, 3, 3, 64, 64, 0);
    packw(erw1 + (long)i*64*64,   WeR1[i], 64, 64, 1, 1, 64, 64, 0);
    packw(drw3 + (long)i*64*64*9, WdR3[i], 64, 64, 3, 3, 64, 64, 0);
    packw(drw1 + (long)i*64*64,   WdR1[i], 64, 64, 1, 1, 64, 64, 0);
  }
  packw(eow, Weo, 64, 64, 1, 1, 64, 64, 0);
  packw(diw, Wdi, 64, 64, 3, 3, 64, 64, 0);
  packw(dtw1, Wt1, 64, 64, 4, 4, 64, 64, 1);   // flip + transpose
  packw(dtw2, Wt2, 64, 64, 4, 4, 64, 64, 1);
  packw(dtw3, Wt3, 15, 64, 4, 4, 64, 16, 1);
  pack_cb_kernel<<<8, 256, 0, stream>>>(codebook, CBB, CNORM);

  // fused scale/shift per layer
  int ssn = 0;
  auto mkss = [&](const float* bias, const float* bn, int C) -> float* {
    float* p = SS + (ssn++) * 128;
    pack_ss_kernel<<<1, 64, 0, stream>>>(bias, bn, p, p + 64, C, 64);
    return p;
  };
  float* ssE1 = mkss(eb1, enc_bn + 0 * 4 * 64, 64);
  float* ssE2 = mkss(eb2, enc_bn + 1 * 4 * 64, 64);
  float* ssE3 = mkss(eb3, enc_bn + 2 * 4 * 64, 64);
  float *ssER3[4], *ssER1[4], *ssDR3[4], *ssDR1[4];
  for (int i = 0; i < 4; ++i) {
    ssER3[i] = mkss(erb3 + i * 64, erbn + (long)(i * 2 + 0) * 4 * 64, 64);
    ssER1[i] = mkss(erb1 + i * 64, erbn + (long)(i * 2 + 1) * 4 * 64, 64);
    ssDR3[i] = mkss(drb3 + i * 64, drbn + (long)(i * 2 + 0) * 4 * 64, 64);
    ssDR1[i] = mkss(drb1 + i * 64, drbn + (long)(i * 2 + 1) * 4 * 64, 64);
  }
  float* ssEo = mkss(eob, nullptr, 64);
  float* ssDi = mkss(dib, nullptr, 64);
  float* ssT1 = mkss(dtb1, nullptr, 64);
  float* ssT2 = mkss(dtb2, nullptr, 64);
  float* ssT3 = mkss(dtb3, nullptr, 15);

  auto conv = [&](const unsigned short* in, int inHs, int inCp, int cShift,
                  const unsigned short* Wp, const float* ss,
                  unsigned short* outb, int outHs, int outHalo,
                  const unsigned short* resid, float* zef, unsigned short* zeb,
                  const float* xin, float* xrec, float* rs,
                  int H, int W, int Cout, int K, int KW, int stride, int inOff,
                  int flags) {
    int twShift = (W == 256) ? 4 : (W == 128) ? 3 : (W == 64) ? 2 : 1;
    int hShift  = (H == 256) ? 8 : (H == 128) ? 7 : (H == 64) ? 6 : 5;
    int kwMagic = (KW == 4) ? 64 : (KW == 3) ? 86 : 256;
    long waves = (long)32 * H * (W / 16);
    int blocks = (int)((waves + 7) / 8);
    conv_wmma_kernel<<<blocks, 256, 0, stream>>>(
        in, inHs, inCp, cShift, Wp, ss, ss + 64, outb, outHs, outHalo, 64,
        resid, zef, zeb, xin, xrec, rs, H, W, Cout, K, KW, kwMagic, stride,
        inOff, twShift, hShift, flags);
  };

  // ---- encoder ----
  conv(A0, 258, 16, 4, Wc1, ssE1, Z1, 130, 1, nullptr, nullptr, nullptr,
       nullptr, nullptr, nullptr, 128, 128, 64, 256, 4, 2, 0,
       F_RELU_OUT | F_STORE_BF16);
  conv(Z1, 130, 64, 6, Wc2, ssE2, Z2, 66, 1, nullptr, nullptr, nullptr,
       nullptr, nullptr, nullptr, 64, 64, 64, 1024, 4, 2, 0,
       F_RELU_OUT | F_STORE_BF16);
  conv(Z2, 66, 64, 6, Wc3, ssE3, RA, 34, 1, nullptr, nullptr, nullptr,
       nullptr, nullptr, nullptr, 32, 32, 64, 1024, 4, 2, 0, F_STORE_BF16);
  {
    unsigned short* pp[2] = {RA, RB};
    int pi = 0;
    for (int i = 0; i < 4; ++i) {
      unsigned short* X = pp[pi];
      unsigned short* Y = pp[1 - pi];
      conv(X, 34, 64, 6, WeR3[i], ssER3[i], RT, 34, 1, nullptr, nullptr,
           nullptr, nullptr, nullptr, nullptr, 32, 32, 64, 576, 3, 1, 0,
           F_RELU_IN | F_STORE_BF16);
      conv(RT, 34, 64, 6, WeR1[i], ssER1[i], Y, 34, 1, X, nullptr, nullptr,
           nullptr, nullptr, nullptr, 32, 32, 64, 64, 1, 1, 1,
           F_RELU_IN | F_RESID | F_STORE_BF16);
      pi ^= 1;
    }
  }
  conv(RA, 34, 64, 6, Weo, ssEo, nullptr, 1, 0, nullptr, ZEF, ZEB, nullptr,
       nullptr, nullptr, 32, 32, 64, 64, 1, 1, 1, F_STORE_ZE);

  // ---- vector quantizer ----
  vq_argmin_kernel<<<256, 256, 262144, stream>>>(ZEB, CBB, CNORM, IDX);
  vq_gather_kernel<<<8192, 256, 0, stream>>>(IDX, codebook, ZEF, Qb, CNT, VQS);

  // ---- decoder ----
  conv(Qb, 34, 64, 6, Wdi, ssDi, RA, 34, 1, nullptr, nullptr, nullptr,
       nullptr, nullptr, nullptr, 32, 32, 64, 576, 3, 1, 0, F_STORE_BF16);
  {
    unsigned short* pp[2] = {RA, RB};
    int pi = 0;
    for (int i = 0; i < 4; ++i) {
      unsigned short* X = pp[pi];
      unsigned short* Y = pp[1 - pi];
      conv(X, 34, 64, 6, WdR3[i], ssDR3[i], RT, 34, 1, nullptr, nullptr,
           nullptr, nullptr, nullptr, nullptr, 32, 32, 64, 576, 3, 1, 0,
           F_RELU_IN | F_STORE_BF16);
      conv(RT, 34, 64, 6, WdR1[i], ssDR1[i], Y, 34, 1, X, nullptr, nullptr,
           nullptr, nullptr, nullptr, 32, 32, 64, 64, 1, 1, 1,
           F_RELU_IN | F_RESID | F_STORE_BF16);
      pi ^= 1;
    }
  }
  // conv_t chain as stride-1 convs over zero-dilated inputs (halo 2)
  upsample_kernel<<<cdiv((long)32*32*32*8, 256), 256, 0, stream>>>(
      RA, U1, (long)32*32*32*8, 5, 34, 67, 1);
  conv(U1, 67, 64, 6, Wt1, ssT1, V1, 66, 1, nullptr, nullptr, nullptr,
       nullptr, nullptr, nullptr, 64, 64, 64, 1024, 4, 1, 0,
       F_RELU_OUT | F_STORE_BF16);
  upsample_kernel<<<cdiv((long)32*64*64*8, 256), 256, 0, stream>>>(
      V1, U2, (long)32*64*64*8, 6, 66, 131, 0);
  conv(U2, 131, 64, 6, Wt2, ssT2, V2, 130, 1, nullptr, nullptr, nullptr,
       nullptr, nullptr, nullptr, 128, 128, 64, 1024, 4, 1, 0,
       F_RELU_OUT | F_STORE_BF16);
  upsample_kernel<<<cdiv((long)32*128*128*8, 256), 256, 0, stream>>>(
      V2, U3, (long)32*128*128*8, 7, 130, 259, 0);
  conv(U3, 259, 64, 6, Wt3, ssT3, nullptr, 1, 0, nullptr, nullptr, nullptr,
       x, dout, RSM, 256, 256, 16, 1024, 4, 1, 0, F_FINAL);

  finalize_kernel<<<1, 256, 0, stream>>>(CNT, VQS, RSM, dout, 31457280L);
}
